// MultiHeadAttention_55808805044352
// MI455X (gfx1250) — matrix operand order repaired and checked by reading, not compiled, not run on hardware
//
#include <hip/hip_runtime.h>
#include <hip/hip_bf16.h>

typedef __bf16 v16bf __attribute__((ext_vector_type(16)));
typedef float  v8f   __attribute__((ext_vector_type(8)));

#define D_MODEL  1024
#define N_HEADS  16
#define HEAD_DIM 64
#define BATCH    2
#define SEQ      2048
#define M_TOT    (BATCH * SEQ)   // 4096

// ---- WMMA operand index helpers, this code's contiguous k order (wave32, bf16 16x16x32) ----
// A matrix 16x32: lane row = lane&15; kbase = (lane>>4)*16; element e -> K = kbase+e.
__device__ __forceinline__ int a_frag_k(int lane, int e) {
    int kbase = (lane >> 4) << 4;
    return kbase + e;
}
// B matrix 32x16: lane col = lane&15; kbase = (lane>>4)*16; element e -> K = kbase+e.
// C/D 16x16 f32: VGPR r -> row r + 8*(lane>>4), col = lane&15.

// ---------------- TDM (Tensor Data Mover) support ----------------
#if defined(__AMDGCN__) && __has_builtin(__builtin_amdgcn_tensor_load_to_lds) && \
    __has_builtin(__builtin_amdgcn_s_wait_tensorcnt)
#define USE_TDM 1
#else
#define USE_TDM 0
#endif

#if USE_TDM
typedef unsigned int u32x4 __attribute__((ext_vector_type(4)));
typedef int          i32x4 __attribute__((ext_vector_type(4)));
typedef int          i32x8 __attribute__((ext_vector_type(8)));

// 2D tile DMA: global (row-major, stride in elements) -> LDS (packed).
// D# layout per CDNA5 ISA. data_size = 2 bytes (bf16).
// This toolchain exposes the 6-arg builtin: (g0, g1, g2, g3, g_extra, cpol).
__device__ __forceinline__ void tdm_load_2d(unsigned ldsOff, unsigned long long gaddr,
                                            int tileD0, int tileD1,
                                            long long strideElems,
                                            int tensD0, int tensD1) {
    u32x4 g0;
    g0[0] = 1u;                                             // count=1, user mode
    g0[1] = ldsOff;                                         // lds_addr (bytes)
    g0[2] = (unsigned)(gaddr & 0xffffffffull);              // global_addr[31:0]
    g0[3] = (unsigned)((gaddr >> 32) & 0x1ffffffull) | (2u << 30); // addr[56:32]|type=2
    i32x8 g1;
    g1[0] = 0x10000;                                        // data_size=1 (2B); no mask/pad
    g1[1] = (int)(((unsigned)tensD0 & 0xffffu) << 16);      // tensor_dim0 lo16 @ [63:48]
    g1[2] = (int)((((unsigned)tensD0 >> 16) & 0xffffu) |
                  (((unsigned)tensD1 & 0xffffu) << 16));    // dim0 hi | dim1 lo
    g1[3] = (int)((((unsigned)tensD1 >> 16) & 0xffffu) |
                  (((unsigned)tileD0 & 0xffffu) << 16));    // dim1 hi | tile_dim0
    g1[4] = (int)((unsigned)tileD1 & 0xffffu);              // tile_dim1; tile_dim2=0
    g1[5] = (int)(strideElems & 0xffffffffll);              // dim0_stride lo32
    g1[6] = (int)((strideElems >> 32) & 0xffffll);          // dim0_stride hi16; dim1_stride=0
    g1[7] = 0;
    i32x4 z4 = (i32x4){0, 0, 0, 0};                         // D# groups 2/3 unused (2D)
    i32x8 z8 = (i32x8){0, 0, 0, 0, 0, 0, 0, 0};
    __builtin_amdgcn_tensor_load_to_lds(g0, g1, z4, z4, z8, 0);
}

__device__ __forceinline__ unsigned lds_byte_off(const void* p) {
    // generic LDS pointer: addr[31:0] is the LDS byte offset
    return (unsigned)(unsigned long long)(uintptr_t)p;
}
#endif

// ---------------- fp32 -> bf16 cast ----------------
__global__ void cast_f32_bf16(const float* __restrict__ src,
                              __bf16* __restrict__ dst, int n) {
    int i = blockIdx.x * blockDim.x + threadIdx.x;
    if (i < n) dst[i] = (__bf16)src[i];
}

// ---------------- projection GEMM ----------------
// X: [M_TOT, D_MODEL] bf16, W: [D_MODEL(out), D_MODEL(in)] bf16 (torch Linear layout).
// TRANSPOSED=0 -> Yh[b,h,l,d];  TRANSPOSED=1 -> Yh[b,h,d,l] (for V).
template <int TRANSPOSED>
__global__ __launch_bounds__(256)
void proj_gemm_kernel(const __bf16* __restrict__ X,
                      const __bf16* __restrict__ W,
                      const float*  __restrict__ bias,
                      __bf16* __restrict__ Yh) {
    const int lane = threadIdx.x & 31;
    const int wave = threadIdx.x >> 5;
    const int t    = blockIdx.x * 8 + wave;          // wave-tile id
    const int NT   = D_MODEL / 64;                   // 16
    const int tileM = (t / NT) * 16;
    const int tileN = (t % NT) * 64;
    const int ln   = lane & 15;
    const int row  = tileM + ln;
    const int kb16 = (lane >> 4) << 4;

    v8f acc[4] = {};
    for (int k0 = 0; k0 < D_MODEL; k0 += 32) {
        v16bf a;
        #pragma unroll
        for (int e = 0; e < 16; ++e)
            a[e] = X[(size_t)row * D_MODEL + k0 + a_frag_k(lane, e)];
        #pragma unroll
        for (int nt = 0; nt < 4; ++nt) {
            const __bf16* wp = W + (size_t)(tileN + nt * 16 + ln) * D_MODEL + k0 + kb16;
            v16bf bm;
            #pragma unroll
            for (int e = 0; e < 16; ++e) bm[e] = wp[e];
            acc[nt] = __builtin_amdgcn_wmma_f32_16x16x32_bf16(
                false, a, false, bm, (short)0, acc[nt], false, false);
        }
    }
    const int hi = lane >> 4;
    #pragma unroll
    for (int nt = 0; nt < 4; ++nt) {
        #pragma unroll
        for (int r = 0; r < 8; ++r) {
            int m = tileM + r + 8 * hi;
            int n = tileN + nt * 16 + ln;
            float v = acc[nt][r] + bias[n];
            int b = m / SEQ, l = m % SEQ;
            int h = n / HEAD_DIM, d = n % HEAD_DIM;
            if (TRANSPOSED)
                Yh[(((size_t)(b * N_HEADS + h)) * HEAD_DIM + d) * SEQ + l] = (__bf16)v;
            else
                Yh[(((size_t)(b * N_HEADS + h)) * SEQ + l) * HEAD_DIM + d] = (__bf16)v;
        }
    }
}

// ---------------- flash attention ----------------
// Block = 8 waves, all on the same (b,h), 8 adjacent 16-row query tiles.
// K tile (32 keys x 64 d) and V^T tile (64 d x 32 keys) staged in LDS once per
// block, double-buffered via the Tensor Data Mover (TENSORcnt), DMA overlapped
// with WMMA compute.  Qh/Kh: [B,H,L,hd]; Vt: [B,H,hd,L]; mask: [B,L] int.
__global__ __launch_bounds__(256)
void flash_attn_kernel(const __bf16* __restrict__ Qh,
                       const __bf16* __restrict__ Kh,
                       const __bf16* __restrict__ Vt,
                       const int*    __restrict__ mask,
                       __bf16* __restrict__ attn) {
    __shared__ __align__(16) __bf16 Ktile[2][32 * 64];   // [k][d]
    __shared__ __align__(16) __bf16 Vtile[2][64 * 32];   // [d][k]
    __shared__ __align__(16) __bf16 pbuf[8][16 * 32];    // per-wave P tile

    const int lane = threadIdx.x & 31;
    const int wave = threadIdx.x >> 5;
    const int nqb  = SEQ / 128;                          // 16 query blocks
    const int b  = blockIdx.x / (N_HEADS * nqb);
    const int h  = (blockIdx.x / nqb) % N_HEADS;
    const int q0 = (blockIdx.x % nqb) * 128 + wave * 16;
    const int hi = lane >> 4;
    const int ln = lane & 15;
    const int kb16 = hi << 4;

    const size_t baseBH = ((size_t)b * N_HEADS + h) * SEQ;
    const __bf16* Kbh = Kh + baseBH * HEAD_DIM;
    const __bf16* Vbh = Vt + ((size_t)b * N_HEADS + h) * (size_t)HEAD_DIM * SEQ;

    // Q fragments (d = 0..31 and 32..63), fixed for the whole loop
    const __bf16* qp = Qh + (baseBH + q0 + ln) * HEAD_DIM;
    v16bf qf0, qf1;
    #pragma unroll
    for (int e = 0; e < 16; ++e) {
        int ak = a_frag_k(lane, e);
        qf0[e] = qp[ak];
        qf1[e] = qp[32 + ak];
    }

    v8f o[4] = {};
    float mrow[8], lrow[8];
    #pragma unroll
    for (int r = 0; r < 8; ++r) { mrow[r] = -1e30f; lrow[r] = 0.0f; }

    const int NIT = SEQ / 32;

#if USE_TDM
    if (wave == 0) {
        tdm_load_2d(lds_byte_off(&Ktile[0][0]), (unsigned long long)(uintptr_t)Kbh,
                    /*tile*/ 64, 32, /*stride*/ HEAD_DIM, /*tensor*/ 64, 32);
        tdm_load_2d(lds_byte_off(&Vtile[0][0]), (unsigned long long)(uintptr_t)Vbh,
                    /*tile*/ 32, 64, /*stride*/ SEQ, /*tensor*/ 32, 64);
        __builtin_amdgcn_s_wait_tensorcnt(0);
    }
    __syncthreads();
#endif

    for (int it = 0; it < NIT; ++it) {
        const int kt = it * 32;
#if USE_TDM
        const int cur = it & 1;
        const int nxt = cur ^ 1;
        if (wave == 0 && it + 1 < NIT) {   // prefetch next tiles while computing
            tdm_load_2d(lds_byte_off(&Ktile[nxt][0]),
                        (unsigned long long)(uintptr_t)(Kbh + (size_t)(kt + 32) * HEAD_DIM),
                        64, 32, HEAD_DIM, 64, 32);
            tdm_load_2d(lds_byte_off(&Vtile[nxt][0]),
                        (unsigned long long)(uintptr_t)(Vbh + kt + 32),
                        32, 64, SEQ, 32, 64);
        }
#else
        const int cur = 0;
        __syncthreads();   // previous iteration's reads done before overwrite
        {
            // cooperative staging: K tile is 4KB contiguous; V^T tile is 64 rows x 64B
            const uint4* gk = (const uint4*)(Kbh + (size_t)kt * HEAD_DIM);
            ((uint4*)&Ktile[0][0])[threadIdx.x] = gk[threadIdx.x];
            int vrow = threadIdx.x >> 2, vc = threadIdx.x & 3;
            const uint4* gv = (const uint4*)(Vbh + (size_t)vrow * SEQ + kt);
            ((uint4*)&Vtile[0][0])[threadIdx.x] = gv[vc];
        }
        __syncthreads();
#endif

        // ---- S = Q . K^T : two 16x16 score tiles covering 32 keys ----
        v8f s[2];
        #pragma unroll
        for (int j = 0; j < 2; ++j) {
            const __bf16* kp = &Ktile[cur][(j * 16 + ln) * 64 + kb16];
            v16bf kf0, kf1;
            #pragma unroll
            for (int e = 0; e < 16; ++e) {
                kf0[e] = kp[e];
                kf1[e] = kp[32 + e];
            }
            v8f z = {};
            z = __builtin_amdgcn_wmma_f32_16x16x32_bf16(false, qf0, false, kf0,
                                                        (short)0, z, false, false);
            z = __builtin_amdgcn_wmma_f32_16x16x32_bf16(false, qf1, false, kf1,
                                                        (short)0, z, false, false);
            s[j] = z;
        }

        const bool msk0 = mask[b * SEQ + kt + ln] != 0;
        const bool msk1 = mask[b * SEQ + kt + 16 + ln] != 0;

        // ---- online softmax over the 32 new columns ----
        #pragma unroll
        for (int r = 0; r < 8; ++r) {
            float s0 = msk0 ? -1e30f : s[0][r] * 0.125f;   // 1/sqrt(64)
            float s1 = msk1 ? -1e30f : s[1][r] * 0.125f;
            float mloc = fmaxf(s0, s1);
            #pragma unroll
            for (int off = 8; off >= 1; off >>= 1)
                mloc = fmaxf(mloc, __shfl_xor(mloc, off, 32));
            float mnew = fmaxf(mrow[r], mloc);
            float corr = __expf(mrow[r] - mnew);
            float p0 = __expf(s0 - mnew);
            float p1 = __expf(s1 - mnew);
            float ps = p0 + p1;
            #pragma unroll
            for (int off = 8; off >= 1; off >>= 1)
                ps += __shfl_xor(ps, off, 32);
            lrow[r] = lrow[r] * corr + ps;
            mrow[r] = mnew;
            #pragma unroll
            for (int dt = 0; dt < 4; ++dt) o[dt][r] *= corr;
            const int prow = r + 8 * hi;
            pbuf[wave][prow * 32 + ln]      = (__bf16)p0;
            pbuf[wave][prow * 32 + 16 + ln] = (__bf16)p1;
        }

        // reload P as A-fragment (wave-private LDS; DS ops are in-order per wave)
        v16bf pa;
        #pragma unroll
        for (int e = 0; e < 16; ++e)
            pa[e] = pbuf[wave][ln * 32 + a_frag_k(lane, e)];

        // ---- O += P . V : 4 hd-subtiles, V^T rows contiguous over k ----
        #pragma unroll
        for (int dt = 0; dt < 4; ++dt) {
            const __bf16* vp = &Vtile[cur][(dt * 16 + ln) * 32 + kb16];
            v16bf vf;
            #pragma unroll
            for (int e = 0; e < 16; ++e) vf[e] = vp[e];
            o[dt] = __builtin_amdgcn_wmma_f32_16x16x32_bf16(
                false, pa, false, vf, (short)0, o[dt], false, false);
        }

#if USE_TDM
        if (wave == 0) __builtin_amdgcn_s_wait_tensorcnt(0);
        __syncthreads();   // next tiles visible; cur buffer free for overwrite
#endif
    }

    // ---- normalize and write [B,L,D] ----
    #pragma unroll
    for (int r = 0; r < 8; ++r) {
        float inv = 1.0f / lrow[r];
        int m = q0 + r + 8 * hi;
        #pragma unroll
        for (int dt = 0; dt < 4; ++dt) {
            float v = o[dt][r] * inv;
            attn[((size_t)b * SEQ + m) * D_MODEL + h * HEAD_DIM + dt * 16 + ln] = (__bf16)v;
        }
    }
}

// ---------------- output GEMM: out[m,n] = X[m,:] . Wo[n,:] + bo[n]  (fp32 out) ----------------
__global__ __launch_bounds__(256)
void out_gemm_kernel(const __bf16* __restrict__ X,
                     const __bf16* __restrict__ W,
                     const float*  __restrict__ bias,
                     float* __restrict__ Y) {
    const int lane = threadIdx.x & 31;
    const int wave = threadIdx.x >> 5;
    const int t    = blockIdx.x * 8 + wave;
    const int NT   = D_MODEL / 64;
    const int tileM = (t / NT) * 16;
    const int tileN = (t % NT) * 64;
    const int ln   = lane & 15;
    const int row  = tileM + ln;
    const int kb16 = (lane >> 4) << 4;

    v8f acc[4] = {};
    for (int k0 = 0; k0 < D_MODEL; k0 += 32) {
        v16bf a;
        #pragma unroll
        for (int e = 0; e < 16; ++e)
            a[e] = X[(size_t)row * D_MODEL + k0 + a_frag_k(lane, e)];
        #pragma unroll
        for (int nt = 0; nt < 4; ++nt) {
            const __bf16* wp = W + (size_t)(tileN + nt * 16 + ln) * D_MODEL + k0 + kb16;
            v16bf bm;
            #pragma unroll
            for (int e = 0; e < 16; ++e) bm[e] = wp[e];
            acc[nt] = __builtin_amdgcn_wmma_f32_16x16x32_bf16(
                false, a, false, bm, (short)0, acc[nt], false, false);
        }
    }
    const int hi = lane >> 4;
    #pragma unroll
    for (int nt = 0; nt < 4; ++nt) {
        #pragma unroll
        for (int r = 0; r < 8; ++r) {
            int m = tileM + r + 8 * hi;
            int n = tileN + nt * 16 + ln;
            Y[(size_t)m * D_MODEL + n] = acc[nt][r] + bias[n];
        }
    }
}

// ---------------- host launcher ----------------
extern "C" void kernel_launch(void* const* d_in, const int* in_sizes, int n_in,
                              void* d_out, int out_size, void* d_ws, size_t ws_size,
                              hipStream_t stream) {
    const float* q  = (const float*)d_in[0];
    const float* k  = (const float*)d_in[1];
    const float* v  = (const float*)d_in[2];
    const int*   mask = (const int*)d_in[3];
    const float* Wq = (const float*)d_in[4];
    const float* bq = (const float*)d_in[5];
    const float* Wk = (const float*)d_in[6];
    const float* bk = (const float*)d_in[7];
    const float* Wv = (const float*)d_in[8];
    const float* bv = (const float*)d_in[9];
    const float* Wo = (const float*)d_in[10];
    const float* bo = (const float*)d_in[11];

    const int nAct = M_TOT * D_MODEL;        // 4,194,304
    const int nW   = D_MODEL * D_MODEL;      // 1,048,576

    char* ws = (char*)d_ws;
    size_t off = 0;
    auto carve = [&](size_t bytes) {
        void* p = ws + off;
        off += (bytes + 255) & ~(size_t)255;
        return p;
    };
    __bf16* qb   = (__bf16*)carve((size_t)nAct * 2);
    __bf16* kb   = (__bf16*)carve((size_t)nAct * 2);
    __bf16* vb   = (__bf16*)carve((size_t)nAct * 2);
    __bf16* Wqb  = (__bf16*)carve((size_t)nW * 2);
    __bf16* Wkb  = (__bf16*)carve((size_t)nW * 2);
    __bf16* Wvb  = (__bf16*)carve((size_t)nW * 2);
    __bf16* Wob  = (__bf16*)carve((size_t)nW * 2);
    __bf16* Qh   = (__bf16*)carve((size_t)nAct * 2);
    __bf16* Kh   = (__bf16*)carve((size_t)nAct * 2);
    __bf16* Vt   = (__bf16*)carve((size_t)nAct * 2);
    __bf16* attn = (__bf16*)carve((size_t)nAct * 2);

    const int CB = 256;
    cast_f32_bf16<<<(nAct + CB - 1) / CB, CB, 0, stream>>>(q, qb, nAct);
    cast_f32_bf16<<<(nAct + CB - 1) / CB, CB, 0, stream>>>(k, kb, nAct);
    cast_f32_bf16<<<(nAct + CB - 1) / CB, CB, 0, stream>>>(v, vb, nAct);
    cast_f32_bf16<<<(nW + CB - 1) / CB, CB, 0, stream>>>(Wq, Wqb, nW);
    cast_f32_bf16<<<(nW + CB - 1) / CB, CB, 0, stream>>>(Wk, Wkb, nW);
    cast_f32_bf16<<<(nW + CB - 1) / CB, CB, 0, stream>>>(Wv, Wvb, nW);
    cast_f32_bf16<<<(nW + CB - 1) / CB, CB, 0, stream>>>(Wo, Wob, nW);

    // (M/16)*(N/64) = 256*16 = 4096 wave tiles -> 512 blocks of 8 waves
    proj_gemm_kernel<0><<<512, 256, 0, stream>>>(qb, Wqb, bq, Qh);
    proj_gemm_kernel<0><<<512, 256, 0, stream>>>(kb, Wkb, bk, Kh);
    proj_gemm_kernel<1><<<512, 256, 0, stream>>>(vb, Wvb, bv, Vt);   // V -> [B,H,hd,L]

    // B*H*(L/128) = 512 blocks, 8 query tiles each
    flash_attn_kernel<<<512, 256, 0, stream>>>(Qh, Kh, Vt, mask, attn);

    out_gemm_kernel<<<512, 256, 0, stream>>>(attn, Wob, bo, (float*)d_out);
}
